// FlashDeepseekV2DecoderLayer_39204461478529
// MI455X (gfx1250) — matrix operand order repaired and checked by reading, not compiled, not run on hardware
//
#include <hip/hip_runtime.h>
#include <hip/hip_bf16.h>

// ---------------- problem constants (fixed by reference) ----------------
#define T_TOK   2048
#define DMODEL  2048
#define NHEAD   16
#define NOPE_D  128
#define ROPE_D  64
#define QDIM_D  192
#define VDIM_D  128
#define QLORA   1536
#define KVLORA  512
#define NEXP    8
#define FEXP    1408
#define FSH     1408
#define MAXR    5120       // 2*T + per-expert pad headroom (40 tiles of 128)
#define NTILES  40
#define SCALE_ATTN 0.07216878364870322f   // 192^-0.5
#define LOG2E   1.4426950408889634f

typedef __attribute__((ext_vector_type(16))) __bf16 v16bf;
typedef __attribute__((ext_vector_type(8)))  float  v8f;
typedef __attribute__((ext_vector_type(8)))  int    v8i;

// fp32 -> bf16 (round to nearest even)
__device__ __forceinline__ unsigned short f2bf(float f) {
  unsigned int u = __float_as_uint(f);
  unsigned int r = u + 0x7FFFu + ((u >> 16) & 1u);
  return (unsigned short)(r >> 16);
}
__device__ __forceinline__ unsigned int f2bf2(float lo, float hi) {
  return (unsigned int)f2bf(lo) | ((unsigned int)f2bf(hi) << 16);
}
__device__ __forceinline__ float bf2f(unsigned short b) {
  return __uint_as_float(((unsigned int)b) << 16);
}

// ---- WMMA fragment loaders from LDS (wave32 layouts per CDNA5 ISA 7.12.2) ----
// A matrix 16x32 bf16: lane m = lane&15; VGPR r holds K pair at
//   k = (r>>2)*16 + (lane>>4)*8 + (r&3)*2   (pairs merge into ds_load_b128)
__device__ __forceinline__ v16bf lds_frag_a(const unsigned short* lds, int row,
                                            int stride, int half, int koff) {
  v8i u;
#pragma unroll
  for (int r = 0; r < 8; ++r) {
    int k = koff + ((r >> 2) << 4) + (half << 3) + ((r & 3) << 1);
    u[r] = *reinterpret_cast<const int*>(lds + row * stride + k);
  }
  return __builtin_bit_cast(v16bf, u);
}
// B matrix 32x16 bf16: lane n = lane&15; VGPR r holds K pair at
//   k = (r>>2)*16 + half*8 + (r&3)*2    (LDS stored [n][k], k contiguous)
__device__ __forceinline__ v16bf lds_frag_b(const unsigned short* lds, int row,
                                            int stride, int half, int koff) {
  v8i u;
#pragma unroll
  for (int r = 0; r < 8; ++r) {
    int k = koff + ((r >> 2) << 4) + (half << 3) + ((r & 3) << 1);
    u[r] = *reinterpret_cast<const int*>(lds + row * stride + k);
  }
  return __builtin_bit_cast(v16bf, u);
}
__device__ __forceinline__ v8f wmma_bf16(v16bf a, v16bf b, v8f c) {
  return __builtin_amdgcn_wmma_f32_16x16x32_bf16(false, a, false, b, (short)0, c,
                                                 false, false);
}

// ---------------- RMSNorm (fp32 in -> bf16 out) ----------------
__global__ void rmsnorm_bf16_kernel(const float* __restrict__ x,
                                    const float* __restrict__ w,
                                    unsigned short* __restrict__ out,
                                    int cols, int inStride, int outStride) {
  const int row = blockIdx.x;
  const float* xr = x + (size_t)row * inStride;
  float ss = 0.f;
  for (int c = threadIdx.x; c < cols; c += 256) { float v = xr[c]; ss += v * v; }
  __shared__ float red[256];
  red[threadIdx.x] = ss;
  __syncthreads();
  for (int s = 128; s > 0; s >>= 1) {
    if (threadIdx.x < s) red[threadIdx.x] += red[threadIdx.x + s];
    __syncthreads();
  }
  const float inv = rsqrtf(red[0] / (float)cols + 1e-6f);
  for (int c = threadIdx.x; c < cols; c += 256)
    out[(size_t)row * outStride + c] = f2bf(xr[c] * inv * w[c]);
}

// ------ generic WMMA GEMM: C[M,N](fp32) = A(bf16) x W(fp32->bf16) --------------
// Block tile 128(M) x 64(N) x 32(K), 8 waves in 4x2 grid, each wave 32x32
// (2x2 WMMA sub-tiles -> 4 fragment loads per 4 WMMAs).
// WN=false: W is [N,K] row-major (C = A @ W^T).  WN=true: W is [K,N] (C = A @ W).
// GATHER: A row index taken from rowTok.  EPI==1: weighted atomic scatter by token.
template <bool WN, bool GATHER, int EPI>
__global__ __launch_bounds__(256) void gemm_bf16_kernel(
    const unsigned short* __restrict__ A, int lda, const float* __restrict__ W,
    size_t wStride, float* __restrict__ C, int N, int K,
    const int* __restrict__ rowTok, const float* __restrict__ rowW,
    const int* __restrict__ tileExp) {
  const int bm = blockIdx.y * 128;
  const int bn = blockIdx.x * 64;
  const float* Wp = W;
  if (tileExp) {
    const int e = tileExp[blockIdx.y];
    if (e < 0) return;
    Wp += (size_t)e * wStride;
  }
  __shared__ unsigned short As[128 * 40];
  __shared__ unsigned short Bs[64 * 40];
  const int tid = threadIdx.x;
  const int wave = tid >> 5, lane = tid & 31;
  const int wm = wave >> 1, wn = wave & 1;
  const int half = lane >> 4, l16 = lane & 15;

  // hoisted staging coordinates
  const int am0 = tid >> 2, am1 = am0 + 64;        // A rows (2 per thread)
  const int ac = (tid & 3) << 3;                   // A k-chunk (8 bf16 = 16B)
  const int sA0 = GATHER ? rowTok[bm + am0] : (bm + am0);
  const int sA1 = GATHER ? rowTok[bm + am1] : (bm + am1);
  const unsigned short* Ar0 = A + (size_t)sA0 * lda + ac;
  const unsigned short* Ar1 = A + (size_t)sA1 * lda + ac;
  // WT: rows n, 8 float4 chunks per row.  WN: rows k, 16 float4 chunks per row.
  const int wtn0 = tid >> 3, wtn1 = wtn0 + 32, wtc = (tid & 7) << 2;
  const int wnk0 = tid >> 4, wnk1 = wnk0 + 16, wnc = (tid & 15) << 2;

  v8f acc[2][2] = {};
  for (int k0 = 0; k0 < K; k0 += 32) {
    // stage A tile 128x32 bf16 (global_load_b128)
    *reinterpret_cast<uint4*>(&As[am0 * 40 + ac]) =
        *reinterpret_cast<const uint4*>(Ar0 + k0);
    *reinterpret_cast<uint4*>(&As[am1 * 40 + ac]) =
        *reinterpret_cast<const uint4*>(Ar1 + k0);
    // stage W tile -> Bs[n][k] bf16
    if (!WN) {
#pragma unroll
      for (int i = 0; i < 2; ++i) {
        int n = i ? wtn1 : wtn0;
        float4 f = *reinterpret_cast<const float4*>(
            &Wp[(size_t)(bn + n) * K + k0 + wtc]);
        uint2 p; p.x = f2bf2(f.x, f.y); p.y = f2bf2(f.z, f.w);
        *reinterpret_cast<uint2*>(&Bs[n * 40 + wtc]) = p;
      }
      if (k0 + 32 < K) {
        __builtin_prefetch(&Wp[(size_t)(bn + wtn0) * K + k0 + 32 + wtc], 0, 1);
        __builtin_prefetch(Ar0 + k0 + 32, 0, 1);
      }
    } else {
#pragma unroll
      for (int i = 0; i < 2; ++i) {
        int k = i ? wnk1 : wnk0;
        float4 f = *reinterpret_cast<const float4*>(
            &Wp[(size_t)(k0 + k) * N + bn + wnc]);
        Bs[(wnc + 0) * 40 + k] = f2bf(f.x);
        Bs[(wnc + 1) * 40 + k] = f2bf(f.y);
        Bs[(wnc + 2) * 40 + k] = f2bf(f.z);
        Bs[(wnc + 3) * 40 + k] = f2bf(f.w);
      }
      if (k0 + 32 < K) {
        __builtin_prefetch(&Wp[(size_t)(k0 + 32 + wnk0) * N + bn + wnc], 0, 1);
        __builtin_prefetch(Ar0 + k0 + 32, 0, 1);
      }
    }
    __syncthreads();
    v16bf a0 = lds_frag_a(As, wm * 32 + l16, 40, half, 0);
    v16bf a1 = lds_frag_a(As, wm * 32 + 16 + l16, 40, half, 0);
    v16bf b0 = lds_frag_b(Bs, wn * 32 + l16, 40, half, 0);
    v16bf b1 = lds_frag_b(Bs, wn * 32 + 16 + l16, 40, half, 0);
    acc[0][0] = wmma_bf16(a0, b0, acc[0][0]);
    acc[0][1] = wmma_bf16(a0, b1, acc[0][1]);
    acc[1][0] = wmma_bf16(a1, b0, acc[1][0]);
    acc[1][1] = wmma_bf16(a1, b1, acc[1][1]);
    __syncthreads();
  }
#pragma unroll
  for (int mt = 0; mt < 2; ++mt) {
#pragma unroll
    for (int nt = 0; nt < 2; ++nt) {
      const int n = bn + wn * 32 + nt * 16 + l16;
#pragma unroll
      for (int r = 0; r < 8; ++r) {
        int m = bm + wm * 32 + mt * 16 + r + half * 8;
        if (EPI == 0) {
          C[(size_t)m * N + n] = acc[mt][nt][r];
        } else {
          atomicAdd(&C[(size_t)rowTok[m] * N + n], rowW[m] * acc[mt][nt][r]);
        }
      }
    }
  }
}

// ---------------- RoPE + head-major Q/K/V assembly (fp32 -> bf16) ----------------
__global__ void assemble_qkv_kernel(const float* __restrict__ qfull,   // [T,3072]
                                    const float* __restrict__ kvfull,  // [T,4096]
                                    const float* __restrict__ ckv,     // [T,576]
                                    unsigned short* __restrict__ Qh,   // [H][T][192]
                                    unsigned short* __restrict__ Kh,   // [H][T][192]
                                    unsigned short* __restrict__ Vh) { // [H][T][128]
  const int t = blockIdx.x;
  __shared__ float cs[32], sn[32];
  if (threadIdx.x < 32) {
    float freq = powf(10000.0f, -2.0f * (float)threadIdx.x / 64.0f);
    float ang = (float)t * freq;
    cs[threadIdx.x] = cosf(ang);
    sn[threadIdx.x] = sinf(ang);
  }
  __syncthreads();
  for (int i = threadIdx.x; i < NHEAD * QDIM_D; i += 256) {
    int hh = i / QDIM_D, j = i % QDIM_D;
    float v;
    if (j < NOPE_D) {
      v = qfull[(size_t)t * 3072 + hh * QDIM_D + j];
    } else {
      int jj = j - NOPE_D, f = jj & 31;
      const float* qb = qfull + (size_t)t * 3072 + hh * QDIM_D + NOPE_D;
      float x = qb[jj];
      float o = (jj < 32) ? -qb[jj + 32] : qb[jj - 32];
      v = x * cs[f] + o * sn[f];
    }
    Qh[((size_t)hh * T_TOK + t) * QDIM_D + j] = f2bf(v);
  }
  for (int i = threadIdx.x; i < NHEAD * QDIM_D; i += 256) {
    int hh = i / QDIM_D, j = i % QDIM_D;
    float v;
    if (j < NOPE_D) {
      v = kvfull[(size_t)t * 4096 + hh * 256 + j];
    } else {
      int jj = j - NOPE_D, f = jj & 31;
      const float* kp = ckv + (size_t)t * (KVLORA + ROPE_D) + KVLORA;
      float x = kp[jj];
      float o = (jj < 32) ? -kp[jj + 32] : kp[jj - 32];
      v = x * cs[f] + o * sn[f];
    }
    Kh[((size_t)hh * T_TOK + t) * QDIM_D + j] = f2bf(v);
  }
  for (int i = threadIdx.x; i < NHEAD * VDIM_D; i += 256) {
    int hh = i / VDIM_D, j = i % VDIM_D;
    Vh[((size_t)hh * T_TOK + t) * VDIM_D + j] =
        f2bf(kvfull[(size_t)t * 4096 + hh * 256 + NOPE_D + j]);
  }
}

// ---------------- flash attention (causal, online softmax, all-WMMA) ------------
__global__ __launch_bounds__(128) void mla_attn_kernel(
    const unsigned short* __restrict__ Qh, const unsigned short* __restrict__ Kh,
    const unsigned short* __restrict__ Vh, unsigned short* __restrict__ O) {
  const int h = blockIdx.y;
  const int q0 = blockIdx.x * 64;
  __shared__ unsigned short Qs[64 * 200];
  __shared__ unsigned short Ks[32 * 200];
  __shared__ unsigned short Vt[128 * 40];      // transposed: [vdim][key]
  __shared__ unsigned short Ps[4][16 * 40];
  const int tid = threadIdx.x;
  const int wave = tid >> 5, lane = tid & 31;
  const int half = lane >> 4, l16 = lane & 15;

  const unsigned short* Qg = Qh + ((size_t)h * T_TOK + q0) * QDIM_D;
#pragma unroll
  for (int i = 0; i < 12; ++i) {               // 64*192/8 = 1536 uint4
    int idx = tid + i * 128;
    int m = idx / 24, c = (idx % 24) << 3;
    *reinterpret_cast<uint4*>(&Qs[m * 200 + c]) =
        *reinterpret_cast<const uint4*>(Qg + (size_t)m * QDIM_D + c);
  }
  __syncthreads();

  float mrow[8], lrow[8];
  v8f oacc[8] = {};
#pragma unroll
  for (int r = 0; r < 8; ++r) { mrow[r] = -1e30f; lrow[r] = 0.f; }

  const unsigned short* Kg = Kh + (size_t)h * T_TOK * QDIM_D;
  const unsigned short* Vg = Vh + (size_t)h * T_TOK * VDIM_D;
  const int nkt = (q0 + 64) / 32;
  const int qrmax = q0 + wave * 16 + 15;

  for (int kt = 0; kt < nkt; ++kt) {
    const int kb = kt * 32;
#pragma unroll
    for (int i = 0; i < 6; ++i) {              // 32*192/8 = 768 uint4
      int idx = tid + i * 128;
      int m = idx / 24, c = (idx % 24) << 3;
      *reinterpret_cast<uint4*>(&Ks[m * 200 + c]) =
          *reinterpret_cast<const uint4*>(Kg + (size_t)(kb + m) * QDIM_D + c);
    }
#pragma unroll
    for (int i = 0; i < 4; ++i) {              // 32*128/8 = 512 uint4, scatter-T
      int idx = tid + i * 128;
      int m = idx >> 4, c = (idx & 15) << 3;
      uint4 v = *reinterpret_cast<const uint4*>(Vg + (size_t)(kb + m) * VDIM_D + c);
      const unsigned short* pv = reinterpret_cast<const unsigned short*>(&v);
#pragma unroll
      for (int j = 0; j < 8; ++j) Vt[(c + j) * 40 + m] = pv[j];
    }
    __syncthreads();

    if (kb <= qrmax) {                       // causal wave-level skip (uniform)
      v8f s0 = {}, s1 = {};
#pragma unroll
      for (int kd = 0; kd < 6; ++kd) {       // 192 = 6 * 32
        v16bf aq = lds_frag_a(Qs, wave * 16 + l16, 200, half, kd * 32);
        v16bf b0 = lds_frag_b(Ks, l16, 200, half, kd * 32);
        v16bf b1 = lds_frag_b(Ks, 16 + l16, 200, half, kd * 32);
        s0 = wmma_bf16(aq, b0, s0);
        s1 = wmma_bf16(aq, b1, s1);
      }
      float mnew[8];
#pragma unroll
      for (int r = 0; r < 8; ++r) {
        int qr = q0 + wave * 16 + r + half * 8;
        int c0 = kb + l16, c1 = c0 + 16;
        float v0 = (c0 <= qr) ? s0[r] * SCALE_ATTN : -1e30f;
        float v1 = (c1 <= qr) ? s1[r] * SCALE_ATTN : -1e30f;
        s0[r] = v0; s1[r] = v1;
        float mx = fmaxf(v0, v1);
        mx = fmaxf(mx, __shfl_xor(mx, 1, 32));
        mx = fmaxf(mx, __shfl_xor(mx, 2, 32));
        mx = fmaxf(mx, __shfl_xor(mx, 4, 32));
        mx = fmaxf(mx, __shfl_xor(mx, 8, 32));
        mnew[r] = fmaxf(mrow[r], mx);
      }
#pragma unroll
      for (int r = 0; r < 8; ++r) {
        float p0 = exp2f((s0[r] - mnew[r]) * LOG2E);
        float p1 = exp2f((s1[r] - mnew[r]) * LOG2E);
        float rs = p0 + p1;
        rs += __shfl_xor(rs, 1, 32);
        rs += __shfl_xor(rs, 2, 32);
        rs += __shfl_xor(rs, 4, 32);
        rs += __shfl_xor(rs, 8, 32);
        float alpha = exp2f((mrow[r] - mnew[r]) * LOG2E);
        lrow[r] = lrow[r] * alpha + rs;
        mrow[r] = mnew[r];
#pragma unroll
        for (int nt = 0; nt < 8; ++nt) oacc[nt][r] *= alpha;
        int prow = r + half * 8;
        Ps[wave][prow * 40 + l16] = f2bf(p0);
        Ps[wave][prow * 40 + 16 + l16] = f2bf(p1);
      }
      // P (C-layout) bounced through per-wave LDS; DS ops are in-order per wave.
      v16bf ap = lds_frag_a(Ps[wave], l16, 40, half, 0);
#pragma unroll
      for (int nt = 0; nt < 8; ++nt) {
        v16bf bv = lds_frag_b(Vt, nt * 16 + l16, 40, half, 0);
        oacc[nt] = wmma_bf16(ap, bv, oacc[nt]);
      }
    }
    __syncthreads();
  }
#pragma unroll
  for (int r = 0; r < 8; ++r) {
    int qr = q0 + wave * 16 + r + half * 8;
    float inv = 1.0f / lrow[r];
#pragma unroll
    for (int nt = 0; nt < 8; ++nt)
      O[(size_t)qr * (NHEAD * VDIM_D) + h * VDIM_D + nt * 16 + l16] =
          f2bf(oacc[nt][r] * inv);
  }
}

// ---------------- MoE routing ----------------
__global__ void gate_route_kernel(const unsigned short* __restrict__ h2B,
                                  const float* __restrict__ gw,
                                  int* __restrict__ topi, float* __restrict__ topw) {
  const int t = blockIdx.x;
  float p[NEXP] = {};
  for (int d = threadIdx.x; d < DMODEL; d += 256) {
    float hv = bf2f(h2B[(size_t)t * DMODEL + d]);
#pragma unroll
    for (int e = 0; e < NEXP; ++e) p[e] += hv * gw[e * DMODEL + d];
  }
  __shared__ float red[NEXP][256];
#pragma unroll
  for (int e = 0; e < NEXP; ++e) red[e][threadIdx.x] = p[e];
  __syncthreads();
  for (int s = 128; s > 0; s >>= 1) {
    if (threadIdx.x < s)
#pragma unroll
      for (int e = 0; e < NEXP; ++e) red[e][threadIdx.x] += red[e][threadIdx.x + s];
    __syncthreads();
  }
  if (threadIdx.x == 0) {
    float mx = red[0][0];
    for (int e = 1; e < NEXP; ++e) mx = fmaxf(mx, red[e][0]);
    float pr[NEXP], sum = 0.f;
    for (int e = 0; e < NEXP; ++e) { pr[e] = expf(red[e][0] - mx); sum += pr[e]; }
    int i0 = 0;
    for (int e = 1; e < NEXP; ++e) if (pr[e] > pr[i0]) i0 = e;
    int i1 = -1;
    for (int e = 0; e < NEXP; ++e)
      if (e != i0 && (i1 < 0 || pr[e] > pr[i1])) i1 = e;
    float w0 = pr[i0] / sum, w1 = pr[i1] / sum, ws = w0 + w1;
    topi[t * 2] = i0; topi[t * 2 + 1] = i1;
    topw[t * 2] = w0 / ws; topw[t * 2 + 1] = w1 / ws;
  }
}

// deterministic serial list builder: per-expert token lists padded to 128-row tiles
__global__ void route_build_kernel(const int* __restrict__ topi,
                                   const float* __restrict__ topw,
                                   int* __restrict__ rowTok, float* __restrict__ rowW,
                                   int* __restrict__ tileExp) {
  if (threadIdx.x != 0 || blockIdx.x != 0) return;
  int r = 0;
  for (int e = 0; e < NEXP; ++e) {
    int start = r;
    for (int t = 0; t < T_TOK; ++t) {
      if (topi[t * 2] == e)     { rowTok[r] = t; rowW[r] = topw[t * 2];     ++r; }
      if (topi[t * 2 + 1] == e) { rowTok[r] = t; rowW[r] = topw[t * 2 + 1]; ++r; }
    }
    while (r & 127) { rowTok[r] = 0; rowW[r] = 0.0f; ++r; }  // weight-0 pad rows
    for (int tl = start >> 7; tl < (r >> 7); ++tl) tileExp[tl] = e;
  }
  for (int tl = r >> 7; tl < NTILES; ++tl) tileExp[tl] = -1;
}

// ---------------- elementwise ----------------
__global__ void silu_mul_kernel(const float* __restrict__ g,
                                const float* __restrict__ u,
                                unsigned short* __restrict__ out, size_t n) {
  size_t i = (size_t)blockIdx.x * 256 + threadIdx.x;
  if (i < n) {
    float gv = g[i];
    out[i] = f2bf((gv / (1.0f + __expf(-gv))) * u[i]);
  }
}
__global__ void add2_kernel(const float* __restrict__ a, const float* __restrict__ b,
                            float* __restrict__ o, size_t n) {
  size_t i = (size_t)blockIdx.x * 256 + threadIdx.x;
  if (i < n) o[i] = a[i] + b[i];
}
__global__ void add3_kernel(const float* __restrict__ a, const float* __restrict__ b,
                            const float* __restrict__ c, float* __restrict__ o,
                            size_t n) {
  size_t i = (size_t)blockIdx.x * 256 + threadIdx.x;
  if (i < n) o[i] = a[i] + b[i] + c[i];
}

// ================================= host =================================
extern "C" void kernel_launch(void* const* d_in, const int* in_sizes, int n_in,
                              void* d_out, int out_size, void* d_ws, size_t ws_size,
                              hipStream_t stream) {
  (void)in_sizes; (void)n_in; (void)out_size; (void)ws_size;
  const float* hs       = (const float*)d_in[0];
  const float* ln1_w    = (const float*)d_in[1];
  const float* q_a_w    = (const float*)d_in[2];
  const float* q_a_ln   = (const float*)d_in[3];
  const float* q_b_w    = (const float*)d_in[4];
  const float* kv_a_w   = (const float*)d_in[5];
  const float* kv_a_ln  = (const float*)d_in[6];
  const float* kv_b_w   = (const float*)d_in[7];
  const float* o_w      = (const float*)d_in[8];
  const float* ln2_w    = (const float*)d_in[9];
  const float* gate_w   = (const float*)d_in[10];
  const float* w_gate_e = (const float*)d_in[11];
  const float* w_up_e   = (const float*)d_in[12];
  const float* w_down_e = (const float*)d_in[13];
  const float* w_gate_s = (const float*)d_in[14];
  const float* w_up_s   = (const float*)d_in[15];
  const float* w_down_s = (const float*)d_in[16];
  float* out = (float*)d_out;

  size_t off = 0;
  auto carve = [&](size_t bytes) -> void* {
    void* p = (char*)d_ws + off;
    off += (bytes + 255) & ~(size_t)255;
    return p;
  };
  const size_t TD = (size_t)T_TOK * DMODEL;
  float* x1 = (float*)carve(TD * 4);          // persistent across both phases
  const size_t mark = off;

  // ---- phase A: attention (arena) ----
  unsigned short* hA    = (unsigned short*)carve(TD * 2);
  float* qa             = (float*)carve((size_t)T_TOK * QLORA * 4);
  unsigned short* qaB   = (unsigned short*)carve((size_t)T_TOK * QLORA * 2);
  float* qfull          = (float*)carve((size_t)T_TOK * NHEAD * QDIM_D * 4);
  float* ckv            = (float*)carve((size_t)T_TOK * (KVLORA + ROPE_D) * 4);
  unsigned short* ckvB  = (unsigned short*)carve((size_t)T_TOK * KVLORA * 2);
  float* kvfull         = (float*)carve((size_t)T_TOK * NHEAD * 256 * 4);
  unsigned short* Qh    = (unsigned short*)carve((size_t)NHEAD * T_TOK * QDIM_D * 2);
  unsigned short* Kh    = (unsigned short*)carve((size_t)NHEAD * T_TOK * QDIM_D * 2);
  unsigned short* Vh    = (unsigned short*)carve((size_t)NHEAD * T_TOK * VDIM_D * 2);
  unsigned short* attnB = (unsigned short*)carve(TD * 2);
  float* oproj          = (float*)carve(TD * 4);

  rmsnorm_bf16_kernel<<<T_TOK, 256, 0, stream>>>(hs, ln1_w, hA, DMODEL, DMODEL, DMODEL);
  gemm_bf16_kernel<false, false, 0><<<dim3(QLORA / 64, T_TOK / 128), 256, 0, stream>>>(
      hA, DMODEL, q_a_w, 0, qa, QLORA, DMODEL, nullptr, nullptr, nullptr);
  rmsnorm_bf16_kernel<<<T_TOK, 256, 0, stream>>>(qa, q_a_ln, qaB, QLORA, QLORA, QLORA);
  gemm_bf16_kernel<false, false, 0><<<dim3(3072 / 64, T_TOK / 128), 256, 0, stream>>>(
      qaB, QLORA, q_b_w, 0, qfull, 3072, QLORA, nullptr, nullptr, nullptr);
  gemm_bf16_kernel<false, false, 0><<<dim3(576 / 64, T_TOK / 128), 256, 0, stream>>>(
      hA, DMODEL, kv_a_w, 0, ckv, 576, DMODEL, nullptr, nullptr, nullptr);
  rmsnorm_bf16_kernel<<<T_TOK, 256, 0, stream>>>(ckv, kv_a_ln, ckvB, KVLORA, 576, KVLORA);
  gemm_bf16_kernel<false, false, 0><<<dim3(4096 / 64, T_TOK / 128), 256, 0, stream>>>(
      ckvB, KVLORA, kv_b_w, 0, kvfull, 4096, KVLORA, nullptr, nullptr, nullptr);
  assemble_qkv_kernel<<<T_TOK, 256, 0, stream>>>(qfull, kvfull, ckv, Qh, Kh, Vh);
  mla_attn_kernel<<<dim3(T_TOK / 64, NHEAD), 128, 0, stream>>>(Qh, Kh, Vh, attnB);
  gemm_bf16_kernel<false, false, 0><<<dim3(DMODEL / 64, T_TOK / 128), 256, 0, stream>>>(
      attnB, NHEAD * VDIM_D, o_w, 0, oproj, DMODEL, NHEAD * VDIM_D, nullptr, nullptr,
      nullptr);
  add2_kernel<<<(TD + 255) / 256, 256, 0, stream>>>(hs, oproj, x1, TD);

  // ---- phase B: MoE (reuse arena) ----
  off = mark;
  unsigned short* h2B = (unsigned short*)carve(TD * 2);
  int* topi           = (int*)carve((size_t)T_TOK * 2 * 4);
  float* topwv        = (float*)carve((size_t)T_TOK * 2 * 4);
  int* rowTok         = (int*)carve((size_t)MAXR * 4);
  float* rowWv        = (float*)carve((size_t)MAXR * 4);
  int* tileExp        = (int*)carve((size_t)NTILES * 4);
  float* g_e          = (float*)carve((size_t)MAXR * FEXP * 4);
  float* u_e          = (float*)carve((size_t)MAXR * FEXP * 4);
  unsigned short* gu  = (unsigned short*)carve((size_t)MAXR * FEXP * 2);
  float* y            = (float*)carve(TD * 4);
  float* gs           = (float*)carve((size_t)T_TOK * FSH * 4);
  float* us           = (float*)carve((size_t)T_TOK * FSH * 4);
  unsigned short* guS = (unsigned short*)carve((size_t)T_TOK * FSH * 2);
  float* ys           = (float*)carve(TD * 4);

  rmsnorm_bf16_kernel<<<T_TOK, 256, 0, stream>>>(x1, ln2_w, h2B, DMODEL, DMODEL, DMODEL);
  gate_route_kernel<<<T_TOK, 256, 0, stream>>>(h2B, gate_w, topi, topwv);
  route_build_kernel<<<1, 1, 0, stream>>>(topi, topwv, rowTok, rowWv, tileExp);
  hipMemsetAsync(y, 0, TD * 4, stream);

  // top-2 experts only: gathered rows, per-tile expert weights
  gemm_bf16_kernel<true, true, 0><<<dim3(FEXP / 64, NTILES), 256, 0, stream>>>(
      h2B, DMODEL, w_gate_e, (size_t)DMODEL * FEXP, g_e, FEXP, DMODEL, rowTok,
      nullptr, tileExp);
  gemm_bf16_kernel<true, true, 0><<<dim3(FEXP / 64, NTILES), 256, 0, stream>>>(
      h2B, DMODEL, w_up_e, (size_t)DMODEL * FEXP, u_e, FEXP, DMODEL, rowTok, nullptr,
      tileExp);
  silu_mul_kernel<<<((size_t)MAXR * FEXP + 255) / 256, 256, 0, stream>>>(
      g_e, u_e, gu, (size_t)MAXR * FEXP);
  gemm_bf16_kernel<true, false, 1><<<dim3(DMODEL / 64, NTILES), 256, 0, stream>>>(
      gu, FEXP, w_down_e, (size_t)FEXP * DMODEL, y, DMODEL, FEXP, rowTok, rowWv,
      tileExp);

  // shared expert
  gemm_bf16_kernel<true, false, 0><<<dim3(FSH / 64, T_TOK / 128), 256, 0, stream>>>(
      h2B, DMODEL, w_gate_s, 0, gs, FSH, DMODEL, nullptr, nullptr, nullptr);
  gemm_bf16_kernel<true, false, 0><<<dim3(FSH / 64, T_TOK / 128), 256, 0, stream>>>(
      h2B, DMODEL, w_up_s, 0, us, FSH, DMODEL, nullptr, nullptr, nullptr);
  silu_mul_kernel<<<((size_t)T_TOK * FSH + 255) / 256, 256, 0, stream>>>(
      gs, us, guS, (size_t)T_TOK * FSH);
  gemm_bf16_kernel<true, false, 0><<<dim3(DMODEL / 64, T_TOK / 128), 256, 0, stream>>>(
      guS, FSH, w_down_s, 0, ys, DMODEL, FSH, nullptr, nullptr, nullptr);

  add3_kernel<<<(TD + 255) / 256, 256, 0, stream>>>(x1, y, ys, out, TD);
}
